// BRITS_60370060313244
// MI455X (gfx1250) — matrix operand order repaired and checked by reading, not compiled, not run on hardware
//
#include <hip/hip_runtime.h>
#include <hip/hip_bf16.h>

// ---------------------------------------------------------------------------
// BRITS bidirectional imputation scan for gfx1250 (MI455X, wave32, WMMA).
//
// Strategy:
//  * Convert all weight matrices to f16 once per launch (workspace, ~3 MB —
//    resident in the 192 MB L2 for the whole scan).
//  * Persistent scan kernel: grid (16 batch-slices) x (2 directions).
//    Each 256-thread workgroup (8 waves) owns 16 batch rows; h/c state and all
//    per-step activations live in LDS (~220 KB, dynamic).  Every GEMM is done
//    with v_wmma_f32_16x16x32_f16: each wave computes 16x16 output tiles over
//    the N dimension of the step's GEMMs.
//  * Decay GEMMs (rr = exp(-relu(lt@lagW^T+b)), rb = rr@rbW^T+b) are computed
//    per step inside the scan (same total FLOPs as precomputing, no 200 MB
//    workspace needed).
// ---------------------------------------------------------------------------

typedef __attribute__((ext_vector_type(16))) _Float16 v16h;
typedef __attribute__((ext_vector_type(8)))  float    v8f;
typedef __attribute__((ext_vector_type(4)))  unsigned int u32x4;

union V16H { v16h v; u32x4 q[2]; };

namespace {
constexpr int    kB  = 256;
constexpr int    kT  = 256;
constexpr int    kF  = 128;
constexpr int    kH  = 256;
constexpr size_t kBTF = (size_t)kB * kT * kF;
// dynamic LDS layout (bytes)
constexpr size_t kSmemF32 = (size_t)16 * (256 + 256 + 256 + 128 + 128 + 128 + 128 + 1024 + 128 + 128) * 4;
constexpr size_t kSmemF16 = (size_t)16 * (256 + 128 + 128 + 256 + 256 + 384 + 256 + 256) * 2;
constexpr size_t kSmemBytes = kSmemF32 + kSmemF16;  // 225,280 B < 320 KB
}

__device__ __forceinline__ float sigmoidf_(float x) { return 1.0f / (1.0f + expf(-x)); }

// A fragment (16x32 f16, M x K) from LDS row-major [16][lda].
// Lane L: M = L&15, half = L>>4; elems 0..15 <- K = kb+half*16.. (contiguous)
__device__ __forceinline__ v16h load_a_frag(const _Float16* A, int lda, int kb, int lane) {
  const int m  = lane & 15;
  const int hf = lane >> 4;
  const _Float16* p0 = A + m * lda + kb + hf * 16;
  V16H r;
  r.q[0] = *(const u32x4*)(p0);
  r.q[1] = *(const u32x4*)(p0 + 8);
  return r.v;
}

// B fragment (32x16 f16, K x N) from a torch-layout weight W[N][K] (row-major).
// Lane L: N = n (caller passes n0 + (L&15)); 16 contiguous K values at kb + (L>>4)*16.
__device__ __forceinline__ v16h load_b_frag(const _Float16* W, int ldb, int n, int kb, int lane) {
  const int kk = kb + ((lane >> 4) << 4);
  const u32x4* q = (const u32x4*)(W + (size_t)n * ldb + kk);
  V16H r;
  r.q[0] = q[0];
  r.q[1] = q[1];
  return r.v;
}

// D/C tile (16x16 f32) -> LDS row-major [16][ldn].
__device__ __forceinline__ void store_d(float* D, int ldn, int n0, v8f d, int lane) {
  const int nn = n0 + (lane & 15);
  const int mb = (lane >> 4) * 8;
#pragma unroll
  for (int r = 0; r < 8; ++r) D[(mb + r) * ldn + nn] = d[r];
}

__device__ __forceinline__ v8f bias_acc(const float* bias, int n) {
  const float bv = bias[n];
  v8f a;
#pragma unroll
  for (int r = 0; r < 8; ++r) a[r] = bv;
  return a;
}

__device__ __forceinline__ v8f bias_acc2(const float* b0, const float* b1, int n) {
  const float bv = b0[n] + b1[n];
  v8f a;
#pragma unroll
  for (int r = 0; r < 8; ++r) a[r] = bv;
  return a;
}

template <int K, int LDA, int LDB>
__device__ __forceinline__ v8f gemm_tile(const _Float16* A, const _Float16* W, int n, v8f acc, int lane) {
#pragma unroll
  for (int kb = 0; kb < K; kb += 32) {
    v16h a = load_a_frag(A, LDA, kb, lane);
    v16h b = load_b_frag(W, LDB, n, kb, lane);
    acc = __builtin_amdgcn_wmma_f32_16x16x32_f16(false, a, false, b, (short)0, acc, false, false);
  }
  return acc;
}

// ---------------------------------------------------------------------------
// f32 -> f16 weight conversion (optionally zeroing the diagonal for z_W)
// ---------------------------------------------------------------------------
__global__ void cvt_f16_kernel(const float* __restrict__ src, _Float16* __restrict__ dst,
                               int rows, int cols, int mask_diag) {
  const int i = blockIdx.x * blockDim.x + threadIdx.x;
  const int n = rows * cols;
  if (i < n) {
    float v = src[i];
    if (mask_diag && (i / cols) == (i % cols)) v = 0.0f;
    dst[i] = (_Float16)v;
  }
}

// ---------------------------------------------------------------------------
// Persistent bidirectional scan
// ---------------------------------------------------------------------------
struct ScanParams {
  const float* x;
  const float* msk;
  const float* lt;
  const _Float16* Wih[2];
  const _Float16* Whh[2];
  const _Float16* linW[2];
  const _Float16* zW[2];
  const _Float16* betaW[2];
  const _Float16* lagW[2];
  const _Float16* rbW[2];
  const _Float16* testW;
  const float* bih[2];
  const float* bhh[2];
  const float* lin_b[2];
  const float* z_b[2];
  const float* beta_b[2];
  const float* lag_b[2];
  const float* rb_b[2];
  const float* test_b;
  float* out;  // 6 * B*T*F: of, zf, cf, ob, zb, cb
};

__launch_bounds__(256, 1)
__global__ void brits_scan_wmma(ScanParams p) {
  extern __shared__ char smem[];
  // f32 buffers
  float* sh_h    = (float*)smem;            // [16][256] hidden state
  float* sh_c    = sh_h    + 16 * 256;      // [16][256] cell state
  float* sh_rr   = sh_c    + 16 * 256;      // [16][256] decay rr
  float* sh_out  = sh_rr   + 16 * 256;      // [16][128] history regression
  float* sh_z    = sh_out  + 16 * 128;      // [16][128] feature regression
  float* sh_rb   = sh_z    + 16 * 128;      // [16][128] r_beta
  float* sh_be   = sh_rb   + 16 * 128;      // [16][128] beta (after test linear)
  float* sh_g    = sh_be   + 16 * 128;      // [16][1024] LSTM gates
  float* sh_x    = sh_g    + 16 * 1024;     // [16][128] x tile
  float* sh_m    = sh_x    + 16 * 128;      // [16][128] mask tile
  // f16 A-matrix staging buffers
  _Float16* a_h   = (_Float16*)(sh_m + 16 * 128);  // [16][256]
  _Float16* a_lt  = a_h   + 16 * 256;              // [16][128]
  _Float16* a_xc  = a_lt  + 16 * 128;              // [16][128]
  _Float16* a_rr  = a_xc  + 16 * 128;              // [16][256]
  _Float16* a_mrb = a_rr  + 16 * 256;              // [16][256] concat(m, rb)
  _Float16* a_be  = a_mrb + 16 * 256;              // [16][384] sigmoid(beta1)
  _Float16* a_cm  = a_be  + 16 * 384;              // [16][256] concat(c_c, m)
  _Float16* a_hr  = a_cm  + 16 * 256;              // [16][256] h * rr

  const int tid  = threadIdx.x;
  const int lane = tid & 31;
  const int wave = tid >> 5;
  const int b0   = blockIdx.x * 16;
  const int dir  = blockIdx.y;

  const _Float16* Wih   = p.Wih[dir];
  const _Float16* Whh   = p.Whh[dir];
  const _Float16* linW  = p.linW[dir];
  const _Float16* zW    = p.zW[dir];
  const _Float16* betaW = p.betaW[dir];
  const _Float16* lagW  = p.lagW[dir];
  const _Float16* rbW   = p.rbW[dir];
  const _Float16* testW = p.testW;
  const float* bih    = p.bih[dir];
  const float* bhh    = p.bhh[dir];
  const float* lin_b  = p.lin_b[dir];
  const float* z_b    = p.z_b[dir];
  const float* beta_b = p.beta_b[dir];
  const float* lag_b  = p.lag_b[dir];
  const float* rb_b   = p.rb_b[dir];
  const float* test_b = p.test_b;

  float* out_o = p.out + (size_t)(dir * 3 + 0) * kBTF;
  float* out_z = p.out + (size_t)(dir * 3 + 1) * kBTF;
  float* out_c = p.out + (size_t)(dir * 3 + 2) * kBTF;

  for (int i = tid; i < 16 * 256; i += 256) { sh_h[i] = 0.0f; sh_c[i] = 0.0f; }
  __syncthreads();

  for (int step = 0; step < kT; ++step) {
    const int tt = dir ? (kT - 1 - step) : step;

    // ---- Phase A: stage x, m, time_lag, h(f16) ----------------------------
    for (int i = tid; i < 16 * 128; i += 256) {
      const int row = i >> 7, f = i & 127;
      const size_t g = ((size_t)(b0 + row) * kT + tt) * kF + f;
      sh_x[i] = p.x[g];
      sh_m[i] = p.msk[g];
      a_lt[i] = (_Float16)p.lt[g];
    }
    for (int i = tid; i < 16 * 256; i += 256) a_h[i] = (_Float16)sh_h[i];
    __syncthreads();

    // ---- Phase B: rr = exp(-relu(lt@lagW^T+b)) [16,256];  out = h@linW^T+b [16,128]
#pragma unroll
    for (int s = 0; s < 2; ++s) {
      const int n0 = (wave + s * 8) * 16;
      const int n = n0 + (lane & 15);
      v8f acc = bias_acc(lag_b, n);
      acc = gemm_tile<128, 128, 128>(a_lt, lagW, n, acc, lane);
#pragma unroll
      for (int r = 0; r < 8; ++r) acc[r] = expf(-fmaxf(acc[r], 0.0f));
      store_d(sh_rr, 256, n0, acc, lane);
    }
    {
      const int n0 = wave * 16;
      const int n = n0 + (lane & 15);
      v8f acc = bias_acc(lin_b, n);
      acc = gemm_tile<256, 256, 256>(a_h, linW, n, acc, lane);
      store_d(sh_out, 128, n0, acc, lane);
    }
    __syncthreads();

    // ---- Phase C: stage a_rr, a_xc = f16(m*x + (1-m)*out) -----------------
    for (int i = tid; i < 16 * 256; i += 256) a_rr[i] = (_Float16)sh_rr[i];
    for (int i = tid; i < 16 * 128; i += 256) {
      const float mv = sh_m[i];
      a_xc[i] = (_Float16)(mv * sh_x[i] + (1.0f - mv) * sh_out[i]);
    }
    __syncthreads();

    // ---- Phase D: rb = rr@rbW^T+b [16,128];  z = x_c@(zW⊙offdiag)^T+b [16,128]
    {
      const int n0 = wave * 16;
      const int n = n0 + (lane & 15);
      v8f acc = bias_acc(rb_b, n);
      acc = gemm_tile<256, 256, 256>(a_rr, rbW, n, acc, lane);
      store_d(sh_rb, 128, n0, acc, lane);
      v8f accz = bias_acc(z_b, n);
      accz = gemm_tile<128, 128, 128>(a_xc, zW, n, accz, lane);
      store_d(sh_z, 128, n0, accz, lane);
    }
    __syncthreads();

    // ---- Phase E: stage concat(m, rb); write out & z outputs --------------
    for (int i = tid; i < 16 * 128; i += 256) {
      const int row = i >> 7, f = i & 127;
      a_mrb[row * 256 + f]       = (_Float16)sh_m[i];
      a_mrb[row * 256 + 128 + f] = (_Float16)sh_rb[i];
      const size_t g = ((size_t)(b0 + row) * kT + tt) * kF + f;
      out_o[g] = sh_out[i];
      out_z[g] = sh_z[i];
    }
    __syncthreads();

    // ---- Phase F: beta1 = sigmoid(concat(m,rb)@betaW^T+b) -> f16 [16,384] -
#pragma unroll
    for (int s = 0; s < 3; ++s) {
      const int n0 = (wave * 3 + s) * 16;
      const int n = n0 + (lane & 15);
      v8f acc = bias_acc(beta_b, n);
      acc = gemm_tile<256, 256, 256>(a_mrb, betaW, n, acc, lane);
      const int mb = (lane >> 4) * 8;
#pragma unroll
      for (int r = 0; r < 8; ++r)
        a_be[(mb + r) * 384 + n] = (_Float16)sigmoidf_(acc[r]);
    }
    __syncthreads();

    // ---- Phase G: beta = beta1@testW^T + b [16,128] -----------------------
    {
      const int n0 = wave * 16;
      const int n = n0 + (lane & 15);
      v8f acc = bias_acc(test_b, n);
      acc = gemm_tile<384, 384, 384>(a_be, testW, n, acc, lane);
      store_d(sh_be, 128, n0, acc, lane);
    }
    __syncthreads();

    // ---- Phase H: c_val/c_c elementwise; write c_val; stage a_cm, a_hr ----
    for (int i = tid; i < 16 * 128; i += 256) {
      const int row = i >> 7, f = i & 127;
      const float be = sh_be[i];
      const float cv = be * sh_z[i] + (1.0f - be) * sh_out[i];
      const float mv = sh_m[i];
      const float cc = mv * sh_x[i] + (1.0f - mv) * cv;
      a_cm[row * 256 + f]       = (_Float16)cc;
      a_cm[row * 256 + 128 + f] = (_Float16)mv;
      const size_t g = ((size_t)(b0 + row) * kT + tt) * kF + f;
      out_c[g] = cv;
    }
    for (int i = tid; i < 16 * 256; i += 256) a_hr[i] = (_Float16)(sh_h[i] * sh_rr[i]);
    __syncthreads();

    // ---- Phase I: gates = concat(c_c,m)@Wih^T + (h*rr)@Whh^T + bih + bhh --
#pragma unroll
    for (int s = 0; s < 8; ++s) {
      const int n0 = (wave * 8 + s) * 16;
      const int n = n0 + (lane & 15);
      v8f acc = bias_acc2(bih, bhh, n);
      acc = gemm_tile<256, 256, 256>(a_cm, Wih, n, acc, lane);
      acc = gemm_tile<256, 256, 256>(a_hr, Whh, n, acc, lane);
      store_d(sh_g, 1024, n0, acc, lane);
    }
    __syncthreads();

    // ---- Phase J: LSTM cell update ---------------------------------------
    for (int i = tid; i < 16 * 256; i += 256) {
      const int row = i >> 8, j = i & 255;
      const int base = row * 1024;
      const float gi = sh_g[base + j];
      const float gf = sh_g[base + 256 + j];
      const float gg = sh_g[base + 512 + j];
      const float go = sh_g[base + 768 + j];
      const float c2 = sigmoidf_(gf) * sh_c[i] + sigmoidf_(gi) * tanhf(gg);
      sh_c[i] = c2;
      sh_h[i] = sigmoidf_(go) * tanhf(c2);
    }
    __syncthreads();
  }
}

// ---------------------------------------------------------------------------
// Host launch
// ---------------------------------------------------------------------------
extern "C" void kernel_launch(void* const* d_in, const int* in_sizes, int n_in,
                              void* d_out, int out_size, void* d_ws, size_t ws_size,
                              hipStream_t stream) {
  (void)in_sizes; (void)n_in; (void)out_size; (void)ws_size;

  // workspace carving (f16, 64B-aligned sizes)
  _Float16* w = (_Float16*)d_ws;
  size_t off = 0;
  auto take = [&](size_t n) -> _Float16* { _Float16* q = w + off; off += n; return q; };
  const size_t SZ_WIH = 1024 * 256, SZ_WHH = 1024 * 256, SZ_LIN = 128 * 256,
               SZ_Z = 128 * 128, SZ_BETA = 384 * 256, SZ_LAG = 256 * 128,
               SZ_RB = 128 * 256, SZ_TEST = 128 * 384;

  _Float16* Wih[2]   = { take(SZ_WIH),  take(SZ_WIH)  };
  _Float16* Whh[2]   = { take(SZ_WHH),  take(SZ_WHH)  };
  _Float16* linW[2]  = { take(SZ_LIN),  take(SZ_LIN)  };
  _Float16* zW[2]    = { take(SZ_Z),    take(SZ_Z)    };
  _Float16* betaW[2] = { take(SZ_BETA), take(SZ_BETA) };
  _Float16* lagW[2]  = { take(SZ_LAG),  take(SZ_LAG)  };
  _Float16* rbW[2]   = { take(SZ_RB),   take(SZ_RB)   };
  _Float16* testW    = take(SZ_TEST);

  struct Cvt { int src; _Float16* dst; int rows, cols, mask; };
  const Cvt cv[] = {
    { 3, Wih[0], 1024, 256, 0 }, { 7, Wih[1], 1024, 256, 0 },
    { 4, Whh[0], 1024, 256, 0 }, { 8, Whh[1], 1024, 256, 0 },
    { 11, linW[0], 128, 256, 0 }, { 13, linW[1], 128, 256, 0 },
    { 15, zW[0], 128, 128, 1 },  { 17, zW[1], 128, 128, 1 },
    { 19, betaW[0], 384, 256, 0 }, { 21, betaW[1], 384, 256, 0 },
    { 23, lagW[0], 256, 128, 0 }, { 25, lagW[1], 256, 128, 0 },
    { 27, rbW[0], 128, 256, 0 }, { 29, rbW[1], 128, 256, 0 },
    { 31, testW, 128, 384, 0 },
  };
  for (const Cvt& c : cv) {
    const int n = c.rows * c.cols;
    cvt_f16_kernel<<<(n + 255) / 256, 256, 0, stream>>>(
        (const float*)d_in[c.src], c.dst, c.rows, c.cols, c.mask);
  }

  ScanParams p{};
  p.x   = (const float*)d_in[0];
  p.msk = (const float*)d_in[1];
  p.lt  = (const float*)d_in[2];
  for (int d = 0; d < 2; ++d) {
    p.Wih[d] = Wih[d];   p.Whh[d] = Whh[d];  p.linW[d] = linW[d];
    p.zW[d] = zW[d];     p.betaW[d] = betaW[d];
    p.lagW[d] = lagW[d]; p.rbW[d] = rbW[d];
  }
  p.testW = testW;
  p.bih[0]    = (const float*)d_in[5];  p.bih[1]    = (const float*)d_in[9];
  p.bhh[0]    = (const float*)d_in[6];  p.bhh[1]    = (const float*)d_in[10];
  p.lin_b[0]  = (const float*)d_in[12]; p.lin_b[1]  = (const float*)d_in[14];
  p.z_b[0]    = (const float*)d_in[16]; p.z_b[1]    = (const float*)d_in[18];
  p.beta_b[0] = (const float*)d_in[20]; p.beta_b[1] = (const float*)d_in[22];
  p.lag_b[0]  = (const float*)d_in[24]; p.lag_b[1]  = (const float*)d_in[26];
  p.rb_b[0]   = (const float*)d_in[28]; p.rb_b[1]   = (const float*)d_in[30];
  p.test_b    = (const float*)d_in[32];
  p.out = (float*)d_out;

  dim3 grid(kB / 16, 2);  // 16 batch slices x 2 directions
  brits_scan_wmma<<<grid, 256, kSmemBytes, stream>>>(p);
}
